// DGCNN_44805098831877
// MI455X (gfx1250) — matrix operand order repaired and checked by reading, not compiled, not run on hardware
//
#include <hip/hip_runtime.h>
#include <hip/hip_bf16.h>

typedef __attribute__((ext_vector_type(16))) _Float16 v16h;
typedef __attribute__((ext_vector_type(8)))  _Float16 v8h;
typedef __attribute__((ext_vector_type(8)))  float    v8f;

#define NPTS 2048
#define NB   8
#define KNN  20

// ---------------------------------------------------------------------------
// Convert f32 features to f16, zero-padded to KPAD channels (WMMA K granularity)
// ---------------------------------------------------------------------------
__global__ void cvt_f16_pad(const float* __restrict__ in, _Float16* __restrict__ out,
                            int C, int KPAD) {
  int p = blockIdx.x;          // point index over B*N
  int c = threadIdx.x;         // 0..KPAD-1
  out[(size_t)p * KPAD + c] = (c < C) ? (_Float16)in[(size_t)p * C + c] : (_Float16)0.0f;
}

// Squared norms from the same f16 data the Gram matrix uses (self-consistent distances)
__global__ void sqnorm_f16(const _Float16* __restrict__ in, float* __restrict__ sq,
                           int KPAD, int total) {
  int p = blockIdx.x * blockDim.x + threadIdx.x;
  if (p >= total) return;
  const _Float16* r = in + (size_t)p * KPAD;
  float s = 0.0f;
  for (int c = 0; c < KPAD; ++c) { float v = (float)r[c]; s += v * v; }
  sq[p] = s;
}

// Split W(2Cin x Cout) into f16 Wd = W_top - W_bot and Wb = W_bot, K-padded to KPAD rows.
// edge feature [xi, xj-xi] @ W  ==  xi@Wd + xj@Wb   (factorized edge-conv)
__global__ void wprep(const float* __restrict__ W, int Cin, int Cout, int KPAD,
                      _Float16* __restrict__ Wd, _Float16* __restrict__ Wb) {
  int i = blockIdx.x * blockDim.x + threadIdx.x;
  if (i >= KPAD * Cout) return;
  int r = i / Cout, c = i % Cout;
  float top = (r < Cin) ? W[(size_t)r * Cout + c]          : 0.0f;
  float bot = (r < Cin) ? W[(size_t)(Cin + r) * Cout + c]  : 0.0f;
  Wd[i] = (_Float16)(top - bot);
  Wb[i] = (_Float16)bot;
}

// ---------------------------------------------------------------------------
// Fused kNN: Gram via v_wmma_f32_16x16x32_f16 with async double-buffered LDS
// staging of column tiles (global_load_async_to_lds_b128 / s_wait_asynccnt),
// then per-row top-20 by successive argmin with wave32 shuffle reductions.
// Block = 128 threads (4 waves), handles 16 rows of one batch.
//
// Stage layout (bank-conflict mitigation for the ds_load_b128 B-fragment reads):
//   KPAD 32/64 : row stride = KPAD+8 halfwords (80/144 B rows -> 16 distinct banks)
//   KPAD 128   : row stride = KPAD, 16B segments XOR-swizzled by row (no LDS growth;
//                strip 128K + stage 32K = 160K exactly)
// ---------------------------------------------------------------------------
template<int KPAD>
__global__ __launch_bounds__(128) void knn_topk(const _Float16* __restrict__ feat,
                                                const float* __restrict__ sq,
                                                int* __restrict__ idxout) {
  constexpr int  CH   = KPAD / 32;             // WMMA K chunks
  constexpr int  SPR  = KPAD / 8;              // 16B segments per feature row
  constexpr int  NSEG = SPR / 2;               // async b128 per lane per 16-row tile
  constexpr bool SWZ  = (KPAD >= 128);
  constexpr int  STRH = SWZ ? KPAD : (KPAD + 8);   // stage row stride in halfwords

  __shared__ float strip[16 * NPTS];               // 16 x 2048 f32 distances (128 KB)
  __shared__ _Float16 stage[4 * 2 * 16 * STRH];    // per-wave double buffers

  const int b    = blockIdx.y;
  const int i0   = blockIdx.x * 16;
  const int lane = threadIdx.x & 31;
  const int wave = threadIdx.x >> 5;
  const int m    = lane & 15;
  const int half = lane >> 4;
  const size_t base = (size_t)b * NPTS;

  // byte offset of 16B segment `seg` of row `row` inside a stage buffer
  auto soff = [](int row, int seg) -> int {
    int s = SWZ ? (seg ^ (row & (SPR - 1))) : seg;
    return row * (STRH * 2) + s * 16;
  };

  // squared norms of my 8 output rows (C-layout rows g + 8*half), in registers
  float sqr[8];
  #pragma unroll
  for (int g = 0; g < 8; ++g) sqr[g] = sq[base + i0 + g + 8 * half];

  // A fragment: rows i0..i0+15, kept in registers for the whole column sweep.
  v16h A[CH];
  const _Float16* arow = feat + (base + i0 + m) * KPAD;
  #pragma unroll
  for (int ch = 0; ch < CH; ++ch) {
    #pragma unroll
    for (int v = 0; v < 8; ++v) {
      int kb = (v & 3) * 2 + (v >> 2) * 16 + half * 8;   // ISA 16-bit A 16x32 layout
      A[ch][2 * v]     = arow[ch * 32 + kb];
      A[ch][2 * v + 1] = arow[ch * 32 + kb + 1];
    }
  }

  _Float16* buf0 = stage + (wave * 2 + 0) * 16 * STRH;
  _Float16* buf1 = stage + (wave * 2 + 1) * 16 * STRH;

  // issue one 16-row column tile into an LDS buffer via async copy (wave-private)
  auto issue = [&](int t, _Float16* buf) {
    const _Float16* src = feat + (base + (size_t)t * 16) * KPAD;
    #pragma unroll
    for (int s = 0; s < NSEG; ++s) {
      int seg  = lane + s * 32;          // global 16B segment id within the tile
      int row  = seg / SPR;
      int sIn  = seg % SPR;
      unsigned ldsa = (unsigned)(uintptr_t)buf + (unsigned)soff(row, sIn);
      const char* ga = (const char*)(src + (size_t)row * KPAD) + sIn * 16;
      asm volatile("global_load_async_to_lds_b128 %0, %1, off"
                   :: "v"(ldsa), "v"(ga) : "memory");
    }
  };

  // Phase 1: each wave sweeps a quarter of the 128 column tiles, double-buffered.
  issue(wave, buf0);
  int bsel = 0;
  for (int t = wave; t < NPTS / 16; t += 4) {
    const bool more = (t + 4) < NPTS / 16;
    if (more) issue(t + 4, bsel ? buf0 : buf1);              // prefetch next tile -> LDS
    if (t + 8 < NPTS / 16)
      __builtin_prefetch(feat + (base + (size_t)(t + 8) * 16) * KPAD, 0, 1); // warm L2
    if (more) asm volatile("s_wait_asynccnt %0" :: "i"(NSEG) : "memory");
    else      asm volatile("s_wait_asynccnt 0" ::: "memory");
    const char* cur = (const char*)(bsel ? buf1 : buf0);

    // two independent accumulator chains (even/odd K chunks) for the matrix pipe
    v8f acc0 = {}, acc1 = {};
    #pragma unroll
    for (int ch = 0; ch < CH; ++ch) {
      v16h Bf;
      #pragma unroll
      for (int j = 0; j < 2; ++j) {                          // 16B groups along K
        v8h gj = *(const v8h*)(cur + soff(m, ch * 4 + half + 2 * j));
        #pragma unroll
        for (int e = 0; e < 8; ++e) Bf[8 * j + e] = gj[e];   // ISA 16-bit B 32x16 layout
      }
      if (ch & 1)
        acc1 = __builtin_amdgcn_wmma_f32_16x16x32_f16(false, A[ch], false, Bf,
                                                      (short)0, acc1, false, false);
      else
        acc0 = __builtin_amdgcn_wmma_f32_16x16x32_f16(false, A[ch], false, Bf,
                                                      (short)0, acc0, false, false);
    }
    const int j0 = t * 16;
    float sqj = sq[base + j0 + m];
    #pragma unroll
    for (int g = 0; g < 8; ++g) {
      int row = g + 8 * half;                                // ISA f32 C/D 16x16 layout
      strip[row * NPTS + j0 + m] = sqr[g] + sqj - 2.0f * (acc0[g] + acc1[g]);
    }
    bsel ^= 1;
  }
  __syncthreads();

  // exclude self once per row (reference adds 1e10 on the diagonal)
  if (threadIdx.x < 16)
    strip[threadIdx.x * NPTS + i0 + threadIdx.x] += 1.0e10f;
  __syncthreads();

  // Phase 2: per-row top-KNN by successive argmin (wave32 shuffle reduce).
  for (int r = wave; r < 16; r += 4) {
    float* srow = strip + r * NPTS;
    int*   orow = idxout + (base + i0 + r) * KNN;
    for (int p = 0; p < KNN; ++p) {
      float best = 3.0e38f; int bj = NPTS;
      for (int j = lane; j < NPTS; j += 32) {
        float v = srow[j];
        if (v < best || (v == best && j < bj)) { best = v; bj = j; }
      }
      #pragma unroll
      for (int off = 16; off > 0; off >>= 1) {
        float ob = __shfl_xor(best, off);
        int   oj = __shfl_xor(bj, off);
        if (ob < best || (ob == best && oj < bj)) { best = ob; bj = oj; }
      }
      if (lane == 0) { orow[p] = bj; srow[bj] = 3.0e38f; }
    }
  }
}

// ---------------------------------------------------------------------------
// Dense GEMM: out(B,N,COUT) f32 = A16(B,N,KPAD) @ W16(KPAD,COUT) [+ bias]
// One wave per 16x64 output tile: 4 independent accumulators share one A
// fragment (4x A reuse, 4-deep independent WMMA chains for the matrix pipe).
// ---------------------------------------------------------------------------
template<int KPAD>
__global__ __launch_bounds__(32) void gemm16(const _Float16* __restrict__ A,
                                             const _Float16* __restrict__ W,
                                             const float* __restrict__ bias, int useBias,
                                             int COUT, float* __restrict__ out) {
  const int b    = blockIdx.z;
  const int i0   = blockIdx.x * 16;
  const int c0   = blockIdx.y * 64;
  const int lane = threadIdx.x;
  const int m    = lane & 15;
  const int half = lane >> 4;

  v8f acc[4] = {};
  const _Float16* arow = A + ((size_t)b * NPTS + i0 + m) * KPAD;
  #pragma unroll
  for (int k0 = 0; k0 < KPAD; k0 += 32) {
    v16h af;
    #pragma unroll
    for (int v = 0; v < 8; ++v) {
      int ka = 2 * v + 16 * half;
      af[2 * v]     = arow[k0 + ka];
      af[2 * v + 1] = arow[k0 + ka + 1];
    }
    #pragma unroll
    for (int cc = 0; cc < 4; ++cc) {
      v16h bf;
      #pragma unroll
      for (int v = 0; v < 8; ++v) {
        int kb = 2 * v + 16 * half;
        bf[2 * v]     = W[(size_t)(k0 + kb)     * COUT + c0 + cc * 16 + m];
        bf[2 * v + 1] = W[(size_t)(k0 + kb + 1) * COUT + c0 + cc * 16 + m];
      }
      acc[cc] = __builtin_amdgcn_wmma_f32_16x16x32_f16(false, af, false, bf,
                                                       (short)0, acc[cc], false, false);
    }
  }
  float* ob = out + (size_t)b * NPTS * COUT;
  #pragma unroll
  for (int cc = 0; cc < 4; ++cc) {
    float bv = useBias ? bias[c0 + cc * 16 + m] : 0.0f;
    #pragma unroll
    for (int g = 0; g < 8; ++g) {
      int row = i0 + g + 8 * half;
      ob[(size_t)row * COUT + c0 + cc * 16 + m] = acc[cc][g] + bv;
    }
  }
}

// out[n] = Q[n] + max_k P[idx[n,k]]   (gather-max; P is L2-resident)
__global__ void combine_max(const float* __restrict__ Q, const float* __restrict__ P,
                            const int* __restrict__ idx, int COUT,
                            float* __restrict__ out) {
  __shared__ int nb[KNN];
  int p = blockIdx.x;                 // point over B*N
  int c = threadIdx.x;                // 0..COUT-1
  if (c < KNN) nb[c] = idx[(size_t)p * KNN + c];
  __syncthreads();
  int b = p / NPTS;
  const float* Pb = P + (size_t)b * NPTS * COUT;
  float mx = -3.0e38f;
  #pragma unroll 4
  for (int kk = 0; kk < KNN; ++kk)
    mx = fmaxf(mx, Pb[(size_t)nb[kk] * COUT + c]);
  out[(size_t)p * COUT + c] = Q[(size_t)p * COUT + c] + mx;
}

// global max pool over N: g(b,c) = max_n h(b,n,c)
__global__ void pool_max(const float* __restrict__ h, float* __restrict__ g) {
  int b = blockIdx.x, c = threadIdx.x;   // 256 threads
  const float* hb = h + (size_t)b * NPTS * 256;
  float mx = -3.0e38f;
  for (int n = 0; n < NPTS; ++n) mx = fmaxf(mx, hb[(size_t)n * 256 + c]);
  g[b * 256 + c] = mx;
}

// tiny fully-connected: out(b,o) = [relu](in(b,:) @ W + bias)
__global__ void fc(const float* __restrict__ in, const float* __restrict__ W,
                   const float* __restrict__ bias, int Cin, int Cout, int doRelu,
                   float* __restrict__ out) {
  int o = blockIdx.x * blockDim.x + threadIdx.x;
  int b = blockIdx.y;
  if (o >= Cout) return;
  float s = bias[o];
  for (int i = 0; i < Cin; ++i) s += in[b * Cin + i] * W[(size_t)i * Cout + o];
  if (doRelu) s = fmaxf(s, 0.0f);
  out[b * Cout + o] = s;
}

// ---------------------------------------------------------------------------
extern "C" void kernel_launch(void* const* d_in, const int* in_sizes, int n_in,
                              void* d_out, int out_size, void* d_ws, size_t ws_size,
                              hipStream_t stream) {
  (void)in_sizes; (void)n_in; (void)out_size; (void)ws_size;
  const float* x    = (const float*)d_in[0];
  /* d_in[1] = k (==20, hardcoded) */
  const float* W1   = (const float*)d_in[2];
  const float* b1   = (const float*)d_in[3];
  const float* W2   = (const float*)d_in[4];
  const float* b2   = (const float*)d_in[5];
  const float* W3   = (const float*)d_in[6];
  const float* b3   = (const float*)d_in[7];
  const float* fc1w = (const float*)d_in[8];
  const float* fc1b = (const float*)d_in[9];
  const float* fc2w = (const float*)d_in[10];
  const float* fc2b = (const float*)d_in[11];
  const float* fc3w = (const float*)d_in[12];
  const float* fc3b = (const float*)d_in[13];
  float* out = (float*)d_out;

  const int total = NB * NPTS;
  char* w = (char*)d_ws;
  size_t off = 0;
  auto alloc = [&](size_t bytes) {
    off = (off + 255) & ~(size_t)255;
    void* p = w + off; off += bytes; return p;
  };
  _Float16* f16a = (_Float16*)alloc((size_t)total * 128 * 2); // features, reused (max KPAD=128)
  float*    sq   = (float*)   alloc((size_t)total * 4);
  int*      idx  = (int*)     alloc((size_t)total * KNN * 4);
  _Float16* Wd   = (_Float16*)alloc((size_t)128 * 256 * 2);
  _Float16* Wb   = (_Float16*)alloc((size_t)128 * 256 * 2);
  float*    P    = (float*)   alloc((size_t)total * 256 * 4);
  float*    Q    = (float*)   alloc((size_t)total * 256 * 4);
  float*    h1   = (float*)   alloc((size_t)total * 64 * 4);
  float*    h2   = (float*)   alloc((size_t)total * 128 * 4);
  float*    h3   = (float*)   alloc((size_t)total * 256 * 4);
  float*    gp   = (float*)   alloc((size_t)NB * 256 * 4);
  float*    f1   = (float*)   alloc((size_t)NB * 512 * 4);
  float*    f2   = (float*)   alloc((size_t)NB * 256 * 4);

  // ---- layer 1: Cin=3 (pad 32) -> Cout=64
  cvt_f16_pad<<<total, 32, 0, stream>>>(x, f16a, 3, 32);
  sqnorm_f16<<<(total + 255) / 256, 256, 0, stream>>>(f16a, sq, 32, total);
  wprep<<<(32 * 64 + 255) / 256, 256, 0, stream>>>(W1, 3, 64, 32, Wd, Wb);
  knn_topk<32><<<dim3(NPTS / 16, NB), 128, 0, stream>>>(f16a, sq, idx);
  gemm16<32><<<dim3(NPTS / 16, 64 / 64, NB), 32, 0, stream>>>(f16a, Wb, nullptr, 0, 64, P);
  gemm16<32><<<dim3(NPTS / 16, 64 / 64, NB), 32, 0, stream>>>(f16a, Wd, b1, 1, 64, Q);
  combine_max<<<total, 64, 0, stream>>>(Q, P, idx, 64, h1);

  // ---- layer 2: Cin=64 -> Cout=128
  cvt_f16_pad<<<total, 64, 0, stream>>>(h1, f16a, 64, 64);
  sqnorm_f16<<<(total + 255) / 256, 256, 0, stream>>>(f16a, sq, 64, total);
  wprep<<<(64 * 128 + 255) / 256, 256, 0, stream>>>(W2, 64, 128, 64, Wd, Wb);
  knn_topk<64><<<dim3(NPTS / 16, NB), 128, 0, stream>>>(f16a, sq, idx);
  gemm16<64><<<dim3(NPTS / 16, 128 / 64, NB), 32, 0, stream>>>(f16a, Wb, nullptr, 0, 128, P);
  gemm16<64><<<dim3(NPTS / 16, 128 / 64, NB), 32, 0, stream>>>(f16a, Wd, b2, 1, 128, Q);
  combine_max<<<total, 128, 0, stream>>>(Q, P, idx, 128, h2);

  // ---- layer 3: Cin=128 -> Cout=256
  cvt_f16_pad<<<total, 128, 0, stream>>>(h2, f16a, 128, 128);
  sqnorm_f16<<<(total + 255) / 256, 256, 0, stream>>>(f16a, sq, 128, total);
  wprep<<<(128 * 256 + 255) / 256, 256, 0, stream>>>(W3, 128, 256, 128, Wd, Wb);
  knn_topk<128><<<dim3(NPTS / 16, NB), 128, 0, stream>>>(f16a, sq, idx);
  gemm16<128><<<dim3(NPTS / 16, 256 / 64, NB), 32, 0, stream>>>(f16a, Wb, nullptr, 0, 256, P);
  gemm16<128><<<dim3(NPTS / 16, 256 / 64, NB), 32, 0, stream>>>(f16a, Wd, b3, 1, 256, Q);
  combine_max<<<total, 256, 0, stream>>>(Q, P, idx, 256, h3);

  // ---- head
  pool_max<<<NB, 256, 0, stream>>>(h3, gp);
  fc<<<dim3((512 + 127) / 128, NB), 128, 0, stream>>>(gp, fc1w, fc1b, 256, 512, 1, f1);
  fc<<<dim3((256 + 127) / 128, NB), 128, 0, stream>>>(f1, fc2w, fc2b, 512, 256, 1, f2);
  fc<<<dim3(1, NB), 128, 0, stream>>>(f2, fc3w, fc3b, 256, 10, 0, out);
}
